// KernelSparseAttention_8435315769779
// MI455X (gfx1250) — matrix operand order repaired and checked by reading, not compiled, not run on hardware
//
#include <hip/hip_runtime.h>
#include <hip/hip_bf16.h>

// ---------------------------------------------------------------------------
// Distance-based attention (B=1, L=2048, D=768, H=12, dk=64) for gfx1250.
// - all GEMMs on v_wmma_f32_16x16x32_bf16 (bf16 in, f32 accumulate)
// - software-pipelined fragment loads (overlap VMEM with WMMA)
// - cooperative K/V tile staging via global_load_async_to_lds_b128
//   (ASYNCcnt + s_wait_asynccnt), double-buffered across the 4 waves/block
// ---------------------------------------------------------------------------

typedef __bf16 bf16;
typedef __attribute__((ext_vector_type(16))) __bf16 v16bf;
typedef __attribute__((ext_vector_type(8)))  __bf16 v8bf;
typedef __attribute__((ext_vector_type(8)))  float  v8f;

#define L_SEQ 2048
#define DIM   768
#define NHEAD 12
#define DK    64

// ---------------- bf16 helpers (round-to-nearest-even) ----------------
static __device__ inline bf16 f2bf(float f) {
  unsigned u = __builtin_bit_cast(unsigned, f);
  u += 0x7FFFu + ((u >> 16) & 1u);
  unsigned short s = (unsigned short)(u >> 16);
  return __builtin_bit_cast(bf16, s);
}
static __device__ inline float bf2f(bf16 b) {
  unsigned short s = __builtin_bit_cast(unsigned short, b);
  unsigned u = ((unsigned)s) << 16;
  return __builtin_bit_cast(float, u);
}

// ---------------- WMMA fragment loaders (contiguous k order: k = 16*(lane/16) + s) --------
// A-matrix 16x32 bf16 (MxK), row-major source, stride lda (elements).
static __device__ inline v16bf load_a_frag(const bf16* __restrict__ A, int lda,
                                           int mBase, int k0, int lane) {
  int row = mBase + (lane & 15);
  int kh  = (lane >> 4) * 16;
  const bf16* p = A + (size_t)row * lda + k0 + kh;
  union { v16bf v; v8bf h[2]; } u;
  u.h[0] = *(const v8bf*)p;
  u.h[1] = *(const v8bf*)(p + 8);
  return u.v;
}

// B-matrix 32x16 bf16 (KxN) where B[k][n] = Bt[n][k], Bt row-major stride ldb.
static __device__ inline v16bf load_bt_frag(const bf16* __restrict__ Bt, int ldb,
                                            int nBase, int k0, int lane) {
  int col = nBase + (lane & 15);
  int kb  = (lane >> 4) * 16;
  const bf16* p = Bt + (size_t)col * ldb + k0 + kb;
  union { v16bf v; v8bf h[2]; } u;
  u.h[0] = *(const v8bf*)p;
  u.h[1] = *(const v8bf*)(p + 8);
  return u.v;
}

// Same B-fragment gather but from an LDS tile (generic ptr -> ds ops).
static __device__ inline v16bf lds_bt_frag(const bf16* tile, int ld, int nOff,
                                           int k0, int lane) {
  const bf16* p = tile + (size_t)(nOff + (lane & 15)) * ld + k0 + ((lane >> 4) * 16);
  union { v16bf v; v8bf h[2]; } u;
  u.h[0] = *(const v8bf*)p;
  u.h[1] = *(const v8bf*)(p + 8);
  return u.v;
}

static __device__ inline v8f wmma_bf16(v16bf a, v16bf b, v8f c) {
  return __builtin_amdgcn_wmma_f32_16x16x32_bf16(false, a, false, b,
                                                 (short)0, c, false, false);
}

// ---------------- async global->LDS copy (16B per lane) ----------------
static __device__ inline void async_copy16(bf16* dstLds, const bf16* srcGlobal) {
  unsigned off = (unsigned)(size_t)dstLds;   // low 32 bits = LDS byte address
  asm volatile("global_load_async_to_lds_b128 %0, %1, off"
               :: "v"(off), "v"(srcGlobal) : "memory");
}
// Stage one 16x64 bf16 K tile (2KB) with 128 lanes: 1 async op per lane.
static __device__ inline void stage_ktile(bf16* dst, const bf16* __restrict__ K,
                                          int jBase, int t) {
  int row = t >> 3, ch = (t & 7) * 8;
  async_copy16(dst + row * DK + ch, K + (size_t)(jBase + row) * DK + ch);
}
// Stage one 64x32 bf16 V tile (4KB) with 128 lanes: 2 async ops per lane.
static __device__ inline void stage_vtile(bf16* dst, const bf16* __restrict__ V,
                                          int jBase, int t) {
#pragma unroll
  for (int q = 0; q < 2; ++q) {
    int idx = t * 2 + q;
    int d = idx >> 2, ch = (idx & 3) * 8;
    async_copy16(dst + d * 32 + ch, V + (size_t)d * L_SEQ + jBase + ch);
  }
}

// ---------------- fp32 -> bf16 conversion kernel ----------------------
__global__ void cvt_f32_bf16_kernel(const float* __restrict__ src,
                                    bf16* __restrict__ dst, int n) {
  int i = blockIdx.x * blockDim.x + threadIdx.x;
  if (i < n) dst[i] = f2bf(src[i]);
}

// ---------------- generic WMMA GEMM:  out = A(MxK) * Bt(NxK)^T + bias --
// mode 0: fp32 row-major [M][N]; mode 1: bf16 [h][M][64]; mode 2: bf16 [h][64][M]
__global__ void gemm_bf16_kernel(const bf16* __restrict__ A,
                                 const bf16* __restrict__ Bt,
                                 const float* __restrict__ bias,
                                 bf16* __restrict__ outB,
                                 float* __restrict__ outF,
                                 int M, int N, int K, int mode) {
  int lane = threadIdx.x & 31;
  int wave = threadIdx.x >> 5;
  int wid  = blockIdx.x * (blockDim.x >> 5) + wave;
  int nTiles = N >> 4;
  int mTile = wid / nTiles;
  int nTile = wid - mTile * nTiles;
  if (mTile >= (M >> 4)) return;   // wave-uniform guard

  int mBase = mTile << 4, nBase = nTile << 4;
  v8f c = {};
  // software pipeline: fragment loads for k+32 overlap the WMMA on k
  v16bf a = load_a_frag(A, K, mBase, 0, lane);
  v16bf b = load_bt_frag(Bt, K, nBase, 0, lane);
  for (int k = 32; k < K; k += 32) {
    __builtin_prefetch(A  + (size_t)(mBase + (lane & 15)) * K + k + 32, 0, 3);
    __builtin_prefetch(Bt + (size_t)(nBase + (lane & 15)) * K + k + 32, 0, 3);
    v16bf an = load_a_frag(A, K, mBase, k, lane);
    v16bf bn = load_bt_frag(Bt, K, nBase, k, lane);
    c = wmma_bf16(a, b, c);
    a = an; b = bn;
  }
  c = wmma_bf16(a, b, c);

  int hi  = lane >> 4;
  int col = nBase + (lane & 15);
  float bv = bias[col];
#pragma unroll
  for (int r = 0; r < 8; ++r) {
    int row = mBase + r + 8 * hi;
    float v = c[r] + bv;
    if (mode == 0) {
      outF[(size_t)row * N + col] = v;
    } else if (mode == 1) {
      int h = col >> 6, d = col & 63;
      outB[((size_t)h * M + row) * DK + d] = f2bf(v);
    } else {
      int h = col >> 6, d = col & 63;
      outB[((size_t)h * DK + d) * M + row] = f2bf(v);
    }
  }
}

// ---------------- per-(head,row) squared norms of bf16 Q/K -------------
__global__ void norms_kernel(const bf16* __restrict__ Qb,
                             const bf16* __restrict__ Kb,
                             float* __restrict__ qn, float* __restrict__ kn) {
  int idx = blockIdx.x * blockDim.x + threadIdx.x;   // h*L + l
  if (idx >= NHEAD * L_SEQ) return;
  const bf16* q = Qb + (size_t)idx * DK;
  const bf16* k = Kb + (size_t)idx * DK;
  float sq = 0.f, sk = 0.f;
#pragma unroll 8
  for (int d = 0; d < DK; ++d) {
    float a = bf2f(q[d]); sq += a * a;
    float b = bf2f(k[d]); sk += b * b;
  }
  qn[idx] = sq; kn[idx] = sk;
}

// ---------------- pass 1: per-head global max of dist_sq ---------------
// grid (32, NHEAD), block 128 (4 waves); wave = one 16-row i-tile.
__global__ void distmax_kernel(const bf16* __restrict__ Qb,
                               const bf16* __restrict__ Kb,
                               const float* __restrict__ qn,
                               const float* __restrict__ kn,
                               unsigned int* __restrict__ dmaxBits) {
  int lane  = threadIdx.x & 31;
  int wave  = threadIdx.x >> 5;
  int iTile = blockIdx.x * 4 + wave;
  int h     = blockIdx.y;
  const bf16* Q = Qb + (size_t)h * L_SEQ * DK;
  const bf16* K = Kb + (size_t)h * L_SEQ * DK;
  int iBase = iTile << 4;
  int hi = lane >> 4, lo = lane & 15;

  v16bf qa0 = load_a_frag(Q, DK, iBase, 0,  lane);
  v16bf qa1 = load_a_frag(Q, DK, iBase, 32, lane);
  float qnv[8];
#pragma unroll
  for (int r = 0; r < 8; ++r) qnv[r] = qn[h * L_SEQ + iBase + r + 8 * hi];

  float m = 0.f;
  v16bf kb0 = load_bt_frag(K, DK, 0, 0,  lane);
  v16bf kb1 = load_bt_frag(K, DK, 0, 32, lane);
  for (int j = 0; j < L_SEQ; j += 16) {
    int jn = (j + 16 < L_SEQ) ? j + 16 : 0;   // pipelined next-tile loads
    v16bf n0 = load_bt_frag(K, DK, jn, 0,  lane);
    v16bf n1 = load_bt_frag(K, DK, jn, 32, lane);
    v8f c = {};
    c = wmma_bf16(qa0, kb0, c);
    c = wmma_bf16(qa1, kb1, c);
    float knv = kn[h * L_SEQ + j + lo];
#pragma unroll
    for (int r = 0; r < 8; ++r) {
      float d = fmaxf(qnv[r] + knv - 2.f * c[r], 0.f) * (1.f / DK);
      m = fmaxf(m, d);
    }
    kb0 = n0; kb1 = n1;
  }
#pragma unroll
  for (int off = 1; off < 32; off <<= 1)
    m = fmaxf(m, __shfl_xor(m, off, 32));
  if (lane == 0) atomicMax(&dmaxBits[h], __float_as_uint(m));  // m >= 0
}

// ---------------- pass 2: row sums + attn-mean + context ---------------
// grid (32, NHEAD), block 128 (4 waves). All 4 waves share head h, so K/V
// tiles are staged once per block into LDS via async copies (double-buffered).
__global__ void attn_kernel(const bf16* __restrict__ Qb,
                            const bf16* __restrict__ Kb,
                            const bf16* __restrict__ Vt,   // [h][64][L]
                            const float* __restrict__ qn,
                            const float* __restrict__ kn,
                            const unsigned int* __restrict__ dmaxBits,
                            const float* __restrict__ theta,
                            float* __restrict__ attnMean,  // [L][L], pre-zeroed
                            bf16* __restrict__ ctxB) {     // [L][DIM]
  __shared__ __attribute__((aligned(16))) bf16 kT [2][16 * DK];  // 2 x 2KB
  __shared__ __attribute__((aligned(16))) bf16 kT2[2][16 * DK];  // 2 x 2KB
  __shared__ __attribute__((aligned(16))) bf16 vT [2][DK * 32];  // 2 x 4KB
  __shared__ unsigned short pstage[4][16 * 36];                  // per-wave P staging

  int t     = threadIdx.x;
  int lane  = t & 31;
  int wave  = t >> 5;
  int iTile = blockIdx.x * 4 + wave;
  int h     = blockIdx.y;
  const bf16* Q = Qb + (size_t)h * L_SEQ * DK;
  const bf16* K = Kb + (size_t)h * L_SEQ * DK;
  const bf16* V = Vt + (size_t)h * DK * L_SEQ;
  int iBase = iTile << 4;
  int hi = lane >> 4, lo = lane & 15;

  // kick off first K tile while we set up
  stage_ktile(kT[0], K, 0, t);

  v16bf qa0 = load_a_frag(Q, DK, iBase, 0,  lane);
  v16bf qa1 = load_a_frag(Q, DK, iBase, 32, lane);
  float qnv[8];
#pragma unroll
  for (int r = 0; r < 8; ++r) qnv[r] = qn[h * L_SEQ + iBase + r + 8 * hi];

  float dmax = __uint_as_float(dmaxBits[h]);
  float ds   = __expf(theta[h]);
  float sc   = ds * (1.f / (dmax + 1e-6f)) * (1.f / DK);

  // ---- pass A: softmax denominators for the 16 owned rows ----
  float rs[8] = {0.f, 0.f, 0.f, 0.f, 0.f, 0.f, 0.f, 0.f};
  {
    int pb = 0;
    for (int j = 0; j < L_SEQ; j += 16, pb ^= 1) {
      int jn = (j + 16 < L_SEQ) ? j + 16 : 0;     // wrap keeps count uniform
      stage_ktile(kT[pb ^ 1], K, jn, t);          // prefetch next tile
      asm volatile("s_wait_asynccnt 0x1" ::: "memory");  // current tile resident
      __syncthreads();
      v16bf b0 = lds_bt_frag(kT[pb], DK, 0, 0,  lane);
      v16bf b1 = lds_bt_frag(kT[pb], DK, 0, 32, lane);
      v8f c = {};
      c = wmma_bf16(qa0, b0, c);
      c = wmma_bf16(qa1, b1, c);
      float knv = kn[h * L_SEQ + j + lo];
#pragma unroll
      for (int r = 0; r < 8; ++r) {
        float d = fmaxf(qnv[r] + knv - 2.f * c[r], 0.f);
        rs[r] += __expf(-sc * d);
      }
      __syncthreads();                            // buffer may be overwritten next
    }
    asm volatile("s_wait_asynccnt 0x0" ::: "memory");
    __syncthreads();
  }
#pragma unroll
  for (int r = 0; r < 8; ++r) {
#pragma unroll
    for (int off = 1; off < 16; off <<= 1)
      rs[r] += __shfl_xor(rs[r], off, 32);   // reduce within each 16-lane half
  }
  float invS[8];
#pragma unroll
  for (int r = 0; r < 8; ++r) invS[r] = 1.f / rs[r];

  // ---- pass B: weights -> attn-mean + P*V context ----
  v8f u0 = {}, u1 = {}, u2 = {}, u3 = {};
  unsigned short* st = &pstage[wave][0];
  const float meanScale = 1.f / NHEAD;

  // preload group for j=0: 2 K tiles + 1 V tile = 4 async ops per lane
  stage_ktile(kT [0], K, 0,  t);
  stage_ktile(kT2[0], K, 16, t);
  stage_vtile(vT [0], V, 0,  t);
  int pb = 0;
  for (int j = 0; j < L_SEQ; j += 32, pb ^= 1) {
    int jn = (j + 32 < L_SEQ) ? j + 32 : 0;
    stage_ktile(kT [pb ^ 1], K, jn,      t);
    stage_ktile(kT2[pb ^ 1], K, jn + 16, t);
    stage_vtile(vT [pb ^ 1], V, jn,      t);
    asm volatile("s_wait_asynccnt 0x4" ::: "memory");  // current group resident
    __syncthreads();

    v8f c0 = {}, c1 = {};
    {
      v16bf b0 = lds_bt_frag(kT[pb], DK, 0, 0,  lane);
      v16bf b1 = lds_bt_frag(kT[pb], DK, 0, 32, lane);
      c0 = wmma_bf16(qa0, b0, c0);
      c0 = wmma_bf16(qa1, b1, c0);
    }
    {
      v16bf b0 = lds_bt_frag(kT2[pb], DK, 0, 0,  lane);
      v16bf b1 = lds_bt_frag(kT2[pb], DK, 0, 32, lane);
      c1 = wmma_bf16(qa0, b0, c1);
      c1 = wmma_bf16(qa1, b1, c1);
    }
    // V B-frags from LDS: B[k][n] = vT[(dBase+n)*32 + k]
    v16bf vb0 = lds_bt_frag(vT[pb], 32, 0,  0, lane);
    v16bf vb1 = lds_bt_frag(vT[pb], 32, 16, 0, lane);
    v16bf vb2 = lds_bt_frag(vT[pb], 32, 32, 0, lane);
    v16bf vb3 = lds_bt_frag(vT[pb], 32, 48, 0, lane);

    float kn0 = kn[h * L_SEQ + j + lo];
    float kn1 = kn[h * L_SEQ + j + 16 + lo];
#pragma unroll
    for (int r = 0; r < 8; ++r) {
      int row = r + 8 * hi;
      float d0 = fmaxf(qnv[r] + kn0 - 2.f * c0[r], 0.f);
      float d1 = fmaxf(qnv[r] + kn1 - 2.f * c1[r], 0.f);
      float w0 = __expf(-sc * d0) * invS[r];
      float w1 = __expf(-sc * d1) * invS[r];
      atomicAdd(&attnMean[(size_t)(iBase + row) * L_SEQ + j + lo],      w0 * meanScale);
      atomicAdd(&attnMean[(size_t)(iBase + row) * L_SEQ + j + 16 + lo], w1 * meanScale);
      st[row * 36 + lo]      = __builtin_bit_cast(unsigned short, f2bf(w0));
      st[row * 36 + 16 + lo] = __builtin_bit_cast(unsigned short, f2bf(w1));
    }
    // LDS-staged C-layout -> A-layout transform; make DS writes visible.
    asm volatile("s_wait_dscnt 0" ::: "memory");
    v16bf pa;
    {
      int row = lo, kh = 16 * hi;
#pragma unroll
      for (int i = 0; i < 8; ++i) {
        pa[i]     = __builtin_bit_cast(bf16, st[row * 36 + kh + i]);
        pa[8 + i] = __builtin_bit_cast(bf16, st[row * 36 + kh + 8 + i]);
      }
    }
    u0 = wmma_bf16(pa, vb0, u0);
    u1 = wmma_bf16(pa, vb1, u1);
    u2 = wmma_bf16(pa, vb2, u2);
    u3 = wmma_bf16(pa, vb3, u3);
    __syncthreads();                      // buffers may be overwritten next iter
  }
  asm volatile("s_wait_asynccnt 0x0" ::: "memory");

  // ---- write context block (bf16, [L][DIM], column h*64 + d) ----
#pragma unroll
  for (int r = 0; r < 8; ++r) {
    int row = iBase + r + 8 * hi;
    bf16* base = ctxB + (size_t)row * DIM + h * DK + lo;
    base[0]  = f2bf(u0[r]);
    base[16] = f2bf(u1[r]);
    base[32] = f2bf(u2[r]);
    base[48] = f2bf(u3[r]);
  }
}

// ---------------------------------------------------------------------------
extern "C" void kernel_launch(void* const* d_in, const int* in_sizes, int n_in,
                              void* d_out, int out_size, void* d_ws, size_t ws_size,
                              hipStream_t stream) {
  (void)in_sizes; (void)n_in; (void)out_size; (void)ws_size;
  const float* x     = (const float*)d_in[0];
  const float* Wq    = (const float*)d_in[1];
  const float* bq    = (const float*)d_in[2];
  const float* Wk    = (const float*)d_in[3];
  const float* bk    = (const float*)d_in[4];
  const float* Wv    = (const float*)d_in[5];
  const float* bv    = (const float*)d_in[6];
  const float* Wo    = (const float*)d_in[7];
  const float* bo    = (const float*)d_in[8];
  const float* theta = (const float*)d_in[9];

  float* out      = (float*)d_out;                 // [L][DIM] fp32
  float* attnMean = out + (size_t)L_SEQ * DIM;     // [L][L]   fp32

  // ---- carve workspace (~21 MB) ----
  size_t off = 0;
  auto carve = [&](size_t bytes) -> void* {
    void* r = (char*)d_ws + off;
    off += (bytes + 255) & ~(size_t)255;
    return r;
  };
  bf16* xb   = (bf16*)carve((size_t)L_SEQ * DIM * 2);
  bf16* wqb  = (bf16*)carve((size_t)DIM * DIM * 2);
  bf16* wkb  = (bf16*)carve((size_t)DIM * DIM * 2);
  bf16* wvb  = (bf16*)carve((size_t)DIM * DIM * 2);
  bf16* wob  = (bf16*)carve((size_t)DIM * DIM * 2);
  bf16* Qb   = (bf16*)carve((size_t)NHEAD * L_SEQ * DK * 2);
  bf16* Kb   = (bf16*)carve((size_t)NHEAD * L_SEQ * DK * 2);
  bf16* Vt   = (bf16*)carve((size_t)NHEAD * DK * L_SEQ * 2);
  bf16* ctxB = (bf16*)carve((size_t)L_SEQ * DIM * 2);
  float* qn  = (float*)carve((size_t)NHEAD * L_SEQ * 4);
  float* kn  = (float*)carve((size_t)NHEAD * L_SEQ * 4);
  unsigned int* dmax = (unsigned int*)carve(64);

  // ---- zero accumulators (graph-capture-safe async memsets) ----
  hipMemsetAsync(attnMean, 0, (size_t)L_SEQ * L_SEQ * 4, stream);
  hipMemsetAsync(dmax, 0, NHEAD * 4, stream);

  // ---- fp32 -> bf16 conversions ----
  {
    int n = L_SEQ * DIM;
    cvt_f32_bf16_kernel<<<(n + 255) / 256, 256, 0, stream>>>(x, xb, n);
    n = DIM * DIM;
    cvt_f32_bf16_kernel<<<(n + 255) / 256, 256, 0, stream>>>(Wq, wqb, n);
    cvt_f32_bf16_kernel<<<(n + 255) / 256, 256, 0, stream>>>(Wk, wkb, n);
    cvt_f32_bf16_kernel<<<(n + 255) / 256, 256, 0, stream>>>(Wv, wvb, n);
    cvt_f32_bf16_kernel<<<(n + 255) / 256, 256, 0, stream>>>(Wo, wob, n);
  }

  // ---- QKV projections: (L/16)*(DIM/16) = 6144 waves -> 768 blocks of 256 ----
  gemm_bf16_kernel<<<768, 256, 0, stream>>>(xb, wqb, bq, Qb, nullptr,
                                            L_SEQ, DIM, DIM, 1);
  gemm_bf16_kernel<<<768, 256, 0, stream>>>(xb, wkb, bk, Kb, nullptr,
                                            L_SEQ, DIM, DIM, 1);
  gemm_bf16_kernel<<<768, 256, 0, stream>>>(xb, wvb, bv, Vt, nullptr,
                                            L_SEQ, DIM, DIM, 2);

  // ---- norms ----
  norms_kernel<<<(NHEAD * L_SEQ + 255) / 256, 256, 0, stream>>>(Qb, Kb, qn, kn);

  // ---- pass 1: global per-head dist max ----
  dim3 ag(32, NHEAD);
  distmax_kernel<<<ag, 128, 0, stream>>>(Qb, Kb, qn, kn, dmax);

  // ---- pass 2: row sums + attn-mean + context ----
  attn_kernel<<<ag, 128, 0, stream>>>(Qb, Kb, Vt, qn, kn, dmax, theta,
                                      attnMean, ctxB);

  // ---- output projection (fp32 out) ----
  gemm_bf16_kernel<<<768, 256, 0, stream>>>(ctxB, wob, bo, nullptr, out,
                                            L_SEQ, DIM, DIM, 0);
}
